// GroupedQueryAttention_80831284511322
// MI455X (gfx1250) — compile-verified, matrix operand order checked by reading, not run on hardware
//
#include <hip/hip_runtime.h>

// ---------------- problem constants ----------------
constexpr int Bc = 2, Tc = 2048, Cc = 2048;
constexpr int Hc = 16, KVc = 4, Dc = 128, Rc = Hc / KVc;
constexpr int Mrows = Bc * Tc;                 // 4096
constexpr int NQKV  = Hc * Dc + 2 * KVc * Dc;  // 3072

typedef __attribute__((ext_vector_type(16))) __bf16 v16bf;
typedef __attribute__((ext_vector_type(8)))  __bf16 v8bf;
typedef __attribute__((ext_vector_type(8)))  float  v8f;

static __device__ __forceinline__ __bf16 f2bf(float f) {
    union { float f; unsigned u; } v; v.f = f;
    unsigned r = v.u + 0x7fffu + ((v.u >> 16) & 1u);   // round-to-nearest-even
    unsigned short h = (unsigned short)(r >> 16);
    return __builtin_bit_cast(__bf16, h);
}

// Load a 16-bit WMMA A/B fragment (16x32) for this lane.
// Lane L owns row (L&15); its 16 K-values are two contiguous runs of 8
// starting at kk + (L>>4)*8 and kk + 16 + (L>>4)*8  -> two b128 loads.
static __device__ __forceinline__ v16bf load_frag(const __bf16* rowptr, int kk, int half) {
    const v8bf lo = *reinterpret_cast<const v8bf*>(rowptr + kk + half * 8);
    const v8bf hi = *reinterpret_cast<const v8bf*>(rowptr + kk + 16 + half * 8);
    return __builtin_shufflevector(lo, hi, 0,1,2,3,4,5,6,7,8,9,10,11,12,13,14,15);
}

// ---------------- elementwise converters ----------------
__global__ __launch_bounds__(256) void conv_f32_bf16(const float* __restrict__ in,
                                                     __bf16* __restrict__ out, int n) {
    int i = blockIdx.x * 256 + threadIdx.x;
    if (i < n) out[i] = f2bf(in[i]);
}

// out[n*Kd + k] = W[k*Nd + n]   (store W^T in bf16)
__global__ __launch_bounds__(256) void tconv_f32_bf16(const float* __restrict__ W,
                                                      __bf16* __restrict__ out,
                                                      int Kd, int Nd) {
    long i = (long)blockIdx.x * 256 + threadIdx.x;
    if (i < (long)Kd * Nd) {
        int n = (int)(i / Kd);
        int k = (int)(i % Kd);
        out[i] = f2bf(W[(size_t)k * Nd + n]);
    }
}

// ---------------- generic bf16 GEMM: C(MxN) = A(MxK) * Bt(NxK)^T ----------------
// One wave computes a 64x16 supertile (4 stacked 16x16 tiles sharing the B fragment).
__global__ __launch_bounds__(256) void gemm_bf16_wmma(const __bf16* __restrict__ A,
                                                      const __bf16* __restrict__ Bt,
                                                      float* __restrict__ Cout,
                                                      int M, int N, int K,
                                                      int lda, int ldb, int ldc) {
    const int lane   = threadIdx.x & 31;
    const int wid    = blockIdx.x * 8 + (threadIdx.x >> 5);
    const int ntiles = N >> 4;
    const int m0     = (wid / ntiles) * 64;
    const int n0     = (wid % ntiles) * 16;
    const int half   = lane >> 4;
    const int l15    = lane & 15;
    if (m0 >= M) return;  // wave-uniform

    const __bf16* brow  = Bt + (size_t)(n0 + l15) * ldb;
    const __bf16* arow0 = A  + (size_t)(m0 +  0 + l15) * lda;
    const __bf16* arow1 = A  + (size_t)(m0 + 16 + l15) * lda;
    const __bf16* arow2 = A  + (size_t)(m0 + 32 + l15) * lda;
    const __bf16* arow3 = A  + (size_t)(m0 + 48 + l15) * lda;

    v8f acc0 = {}, acc1 = {}, acc2 = {}, acc3 = {};
    for (int kk = 0; kk < K; kk += 32) {
        v16bf bfr = load_frag(brow,  kk, half);
        v16bf a0  = load_frag(arow0, kk, half);
        v16bf a1  = load_frag(arow1, kk, half);
        v16bf a2  = load_frag(arow2, kk, half);
        v16bf a3  = load_frag(arow3, kk, half);
        acc0 = __builtin_amdgcn_wmma_f32_16x16x32_bf16(false, a0, false, bfr, (short)0, acc0, false, false);
        acc1 = __builtin_amdgcn_wmma_f32_16x16x32_bf16(false, a1, false, bfr, (short)0, acc1, false, false);
        acc2 = __builtin_amdgcn_wmma_f32_16x16x32_bf16(false, a2, false, bfr, (short)0, acc2, false, false);
        acc3 = __builtin_amdgcn_wmma_f32_16x16x32_bf16(false, a3, false, bfr, (short)0, acc3, false, false);
    }
    const int col = n0 + l15;
#pragma unroll
    for (int i = 0; i < 8; ++i) {
        const int r = i + 8 * half;
        Cout[(size_t)(m0 +  0 + r) * ldc + col] = acc0[i];
        Cout[(size_t)(m0 + 16 + r) * ldc + col] = acc1[i];
        Cout[(size_t)(m0 + 32 + r) * ldc + col] = acc2[i];
        Cout[(size_t)(m0 + 48 + r) * ldc + col] = acc3[i];
    }
}

// ---------------- RoPE + split/scatter to bf16 Q/K (head-major) and V^T ----------------
// V is stored TRANSPOSED per (b, kv-head): Vt[((b*KV+g)*D + d)*T + t], so the
// attention PV stage can fetch B-fragments with contiguous b128 loads.
__global__ __launch_bounds__(256) void rope_split_kernel(const float* __restrict__ qkv,
                                                         __bf16* __restrict__ Qo,
                                                         __bf16* __restrict__ Ko,
                                                         __bf16* __restrict__ Vt) {
    const int row = blockIdx.x;           // b*T + t
    const int b = row / Tc, t = row % Tc;
    const float* src = qkv + (size_t)row * NQKV;
    constexpr float kLnInv = 0.14391156831212787f;  // ln(10000)/64
    for (int idx = threadIdx.x; idx < NQKV; idx += 256) {
        if (idx < Hc * Dc) {                    // Q
            const int hh = idx >> 7, d = idx & 127;
            const float val = src[idx];
            const float other = src[(hh << 7) + ((d < 64) ? (d + 64) : (d - 64))];
            const float rot = (d < 64) ? -other : other;
            const float ang = (float)t * __expf(-(float)(d & 63) * kLnInv);
            float sn, cs; __sincosf(ang, &sn, &cs);
            Qo[(((size_t)(b * Hc + hh)) * Tc + t) * Dc + d] = f2bf(val * cs + rot * sn);
        } else if (idx < Hc * Dc + KVc * Dc) {  // K
            const int j = idx - Hc * Dc;
            const int gg = j >> 7, d = j & 127;
            const float val = src[idx];
            const float other = src[Hc * Dc + (gg << 7) + ((d < 64) ? (d + 64) : (d - 64))];
            const float rot = (d < 64) ? -other : other;
            const float ang = (float)t * __expf(-(float)(d & 63) * kLnInv);
            float sn, cs; __sincosf(ang, &sn, &cs);
            Ko[(((size_t)(b * KVc + gg)) * Tc + t) * Dc + d] = f2bf(val * cs + rot * sn);
        } else {                                // V (no rope) -> transposed layout
            const int j = idx - Hc * Dc - KVc * Dc;
            const int gg = j >> 7, d = j & 127;
            Vt[(((size_t)(b * KVc + gg)) * Dc + d) * Tc + t] = f2bf(src[idx]);
        }
    }
}

// ---------------- flash-style causal GQA attention ----------------
// One wave owns a 16-query tile of one head; iterates key blocks of 32
// with online softmax. 8 WMMAs for Q*K^T + 8 WMMAs for P*V per block.
__global__ __launch_bounds__(256) void gqa_attention_wmma(const __bf16* __restrict__ Q,
                                                          const __bf16* __restrict__ Kt,
                                                          const __bf16* __restrict__ Vt,
                                                          __bf16* __restrict__ Out) {
    __shared__ float pbuf[8][16 * 32];   // per-wave P staging (16KB)
    const int lane = threadIdx.x & 31;
    const int w    = threadIdx.x >> 5;
    const int wid  = blockIdx.x * 8 + w;
    const int qtiles = Tc / 16;
    const int qt = wid % qtiles;
    const int h  = (wid / qtiles) % Hc;
    const int b  = wid / (qtiles * Hc);
    const int g  = h / Rc;
    const int t0 = qt * 16;
    const int half = lane >> 4;
    const int l15  = lane & 15;

    // Q fragments for the whole tile (16 x 128) held in registers.
    const __bf16* qrow = Q + (((size_t)(b * Hc + h)) * Tc + t0 + l15) * Dc;
    v16bf qf[4];
#pragma unroll
    for (int i = 0; i < 4; ++i) qf[i] = load_frag(qrow, i * 32, half);

    const __bf16* Kb  = Kt + ((size_t)(b * KVc + g)) * Tc * Dc;
    // V^T row for this lane's d-column within each 16-wide d-tile.
    const __bf16* vtb = Vt + (((size_t)(b * KVc + g)) * Dc) * Tc;

    v8f o[8] = {};
    float m_i[8], l_i[8];
#pragma unroll
    for (int i = 0; i < 8; ++i) { m_i[i] = -3.0e38f; l_i[i] = 0.f; }

    const float scale = 0.08838834764831845f;  // 1/sqrt(128)
    const int nb = (t0 + 16 + 31) / 32;
    for (int blk = 0; blk < nb; ++blk) {
        const int s0 = blk * 32;
        // prefetch next K block (emits global_prefetch_b8)
        if (blk + 1 < nb)
            __builtin_prefetch(Kb + (size_t)(s0 + 32 + l15) * Dc, 0, 1);
        // ---- S = Q * K^T for 32 keys (two 16-wide n-tiles) ----
        v8f s0v = {}, s1v = {};
#pragma unroll
        for (int nt = 0; nt < 2; ++nt) {
            int krow = s0 + nt * 16 + l15;
            if (krow > Tc - 1) krow = Tc - 1;     // clamp (masked anyway)
            const __bf16* kr = Kb + (size_t)krow * Dc;
#pragma unroll
            for (int ks = 0; ks < 4; ++ks) {
                v16bf bfr = load_frag(kr, ks * 32, half);
                if (nt == 0)
                    s0v = __builtin_amdgcn_wmma_f32_16x16x32_bf16(false, qf[ks], false, bfr, (short)0, s0v, false, false);
                else
                    s1v = __builtin_amdgcn_wmma_f32_16x16x32_bf16(false, qf[ks], false, bfr, (short)0, s1v, false, false);
            }
        }
        // ---- scale, causal mask, online softmax update ----
#pragma unroll
        for (int i = 0; i < 8; ++i) {
            const int trow = t0 + i + 8 * half;
            float v0 = s0v[i] * scale; if (s0 + l15      > trow) v0 = -3.0e38f;
            float v1 = s1v[i] * scale; if (s0 + 16 + l15 > trow) v1 = -3.0e38f;
            float mx = fmaxf(v0, v1);
            mx = fmaxf(mx, __shfl_xor(mx, 1));
            mx = fmaxf(mx, __shfl_xor(mx, 2));
            mx = fmaxf(mx, __shfl_xor(mx, 4));
            mx = fmaxf(mx, __shfl_xor(mx, 8));
            const float mnew  = fmaxf(m_i[i], mx);
            const float alpha = __expf(m_i[i] - mnew);
            const float p0 = __expf(v0 - mnew);
            const float p1 = __expf(v1 - mnew);
            float rs = p0 + p1;
            rs += __shfl_xor(rs, 1);
            rs += __shfl_xor(rs, 2);
            rs += __shfl_xor(rs, 4);
            rs += __shfl_xor(rs, 8);
            l_i[i] = l_i[i] * alpha + rs;
            m_i[i] = mnew;
#pragma unroll
            for (int dt = 0; dt < 8; ++dt) o[dt][i] *= alpha;
            const int prow = i + 8 * half;
            pbuf[w][prow * 32 + l15]      = p0;
            pbuf[w][prow * 32 + 16 + l15] = p1;
        }
        asm volatile("s_wait_dscnt 0" ::: "memory");  // LDS store->load within wave
        // ---- rebuild P as an A-fragment (C-layout -> A-layout transpose) ----
        v16bf pf;
        {
            const float* pr = &pbuf[w][l15 * 32 + half * 8];
#pragma unroll
            for (int e = 0; e < 8; ++e) {
                pf[e]     = f2bf(pr[e]);
                pf[e + 8] = f2bf(pr[16 + e]);
            }
        }
        // ---- O += P * V : B-fragment from V^T rows, contiguous b128 loads ----
        // (tail reads past t=T-1 are multiplied by p==0, harmless and in-bounds
        //  of the workspace allocation)
#pragma unroll
        for (int dt = 0; dt < 8; ++dt) {
            const __bf16* vtrow = vtb + (size_t)(dt * 16 + l15) * Tc;
            v16bf vf = load_frag(vtrow, s0, half);
            o[dt] = __builtin_amdgcn_wmma_f32_16x16x32_bf16(false, pf, false, vf, (short)0, o[dt], false, false);
        }
    }
    // ---- normalize, store to (B*T, C) bf16 for the output projection ----
    float rl[8];
#pragma unroll
    for (int i = 0; i < 8; ++i) rl[i] = __frcp_rn(l_i[i]);
#pragma unroll
    for (int dt = 0; dt < 8; ++dt) {
        const int dcol = dt * 16 + l15;
#pragma unroll
        for (int i = 0; i < 8; ++i) {
            const int trow = t0 + i + 8 * half;
            Out[((size_t)(b * Tc + trow)) * Cc + h * Dc + dcol] = f2bf(o[dt][i] * rl[i]);
        }
    }
}

// ---------------- host-side orchestration ----------------
extern "C" void kernel_launch(void* const* d_in, const int* in_sizes, int n_in,
                              void* d_out, int out_size, void* d_ws, size_t ws_size,
                              hipStream_t stream) {
    const float* x  = (const float*)d_in[0];
    const float* Wq = (const float*)d_in[1];
    const float* Wk = (const float*)d_in[2];
    const float* Wv = (const float*)d_in[3];
    const float* Wo = (const float*)d_in[4];
    float* out = (float*)d_out;

    // workspace carve-up (element counts)
    __bf16* x_bf   = (__bf16*)d_ws;                         // 8M bf16
    __bf16* wqkv_t = x_bf   + (size_t)Mrows * Cc;           // 6M bf16 (3072 x 2048)
    __bf16* wo_t   = wqkv_t + (size_t)NQKV * Cc;            // 4M bf16
    float*  qkv    = (float*)(wo_t + (size_t)Cc * Cc);      // 12M f32
    __bf16* Qb     = (__bf16*)(qkv + (size_t)Mrows * NQKV); // 8M bf16
    __bf16* Kb     = Qb + (size_t)Bc * Hc  * Tc * Dc;       // 2M bf16
    __bf16* Vtb    = Kb + (size_t)Bc * KVc * Tc * Dc;       // 2M bf16 (transposed)
    __bf16* attn   = Vtb + (size_t)Bc * KVc * Tc * Dc;      // 8M bf16

    // 1) convert x, transpose-convert weights (Wq|Wk|Wv concat along N)
    {
        int n = Mrows * Cc;
        conv_f32_bf16<<<(n + 255) / 256, 256, 0, stream>>>(x, x_bf, n);
    }
    tconv_f32_bf16<<<(Cc * (Hc * Dc)  + 255) / 256, 256, 0, stream>>>(Wq, wqkv_t,                        Cc, Hc * Dc);
    tconv_f32_bf16<<<(Cc * (KVc * Dc) + 255) / 256, 256, 0, stream>>>(Wk, wqkv_t + (size_t)Hc * Dc * Cc, Cc, KVc * Dc);
    tconv_f32_bf16<<<(Cc * (KVc * Dc) + 255) / 256, 256, 0, stream>>>(Wv, wqkv_t + (size_t)(Hc + KVc) * Dc * Cc, Cc, KVc * Dc);
    tconv_f32_bf16<<<(Cc * Cc + 255) / 256, 256, 0, stream>>>(Wo, wo_t, Cc, Cc);

    // 2) fused QKV projection: (4096 x 3072) = x_bf * Wqkv^T
    {
        int waves = (Mrows / 64) * (NQKV / 16);   // 12288
        gemm_bf16_wmma<<<waves / 8, 256, 0, stream>>>(x_bf, wqkv_t, qkv,
                                                      Mrows, NQKV, Cc, Cc, Cc, NQKV);
    }
    // 3) RoPE + split into head-major bf16 Q/K and transposed V
    rope_split_kernel<<<Mrows, 256, 0, stream>>>(qkv, Qb, Kb, Vtb);

    // 4) causal GQA flash attention
    {
        int waves = Bc * Hc * (Tc / 16);          // 4096
        gqa_attention_wmma<<<waves / 8, 256, 0, stream>>>(Qb, Kb, Vtb, attn);
    }
    // 5) output projection: out = attn * Wo
    {
        int waves = (Mrows / 64) * (Cc / 16);     // 8192
        gemm_bf16_wmma<<<waves / 8, 256, 0, stream>>>(attn, wo_t, out,
                                                      Mrows, Cc, Cc, Cc, Cc, Cc);
    }
}
